// Mamba2JAX_50629074485313
// MI455X (gfx1250) — matrix operand order repaired and checked by reading, not compiled, not run on hardware
//
#include <hip/hip_runtime.h>
#include <hip/hip_bf16.h>

typedef __attribute__((ext_vector_type(16))) _Float16 v16h;
typedef __attribute__((ext_vector_type(8)))  float    v8f;
typedef int v4i_vec __attribute__((vector_size(16)));
typedef __attribute__((address_space(1))) v4i_vec* gas_v4i;
typedef __attribute__((address_space(3))) v4i_vec* las_v4i;

#define BATCHN 2
#define SEQ    4096
#define DMODEL 2048
#define NH     32
#define HD     128
#define DIN    4096      // EXPAND * D_MODEL
#define CONVD  4224      // D_INNER + 2*D_STATE
#define DIP    8352      // 2*D_INNER + 2*D_STATE + NHEADS
#define DIPP   8384      // DIP padded to multiple of 64 for WMMA tiling
#define NC     16        // SEQ / CHUNK
#define CHK    256
#define NSTATE 64

#if defined(__has_builtin)
#  if __has_builtin(__builtin_amdgcn_global_load_async_to_lds_b128)
#    define HAVE_ASYNC_LDS 1
#  endif
#endif
#ifndef HAVE_ASYNC_LDS
#  define HAVE_ASYNC_LDS 0
#endif

__device__ __forceinline__ v8f vzero8(){ v8f v = {0.f,0.f,0.f,0.f,0.f,0.f,0.f,0.f}; return v; }
__device__ __forceinline__ v8f wmma16(v16h a, v16h b, v8f c){
  return __builtin_amdgcn_wmma_f32_16x16x32_f16(false, a, false, b, (short)0, c, false, false);
}
__device__ __forceinline__ float silu_f(float x){ return x / (1.f + __expf(-x)); }

// 16-byte global -> LDS copy; async (ASYNCcnt) on CDNA5, sync fallback otherwise.
__device__ __forceinline__ void copy16_g2l(const _Float16* g, _Float16* l){
#if HAVE_ASYNC_LDS
  __builtin_amdgcn_global_load_async_to_lds_b128(
      (gas_v4i)(void*)g,
      (las_v4i)(void*)l, 0, 0);
#else
  *reinterpret_cast<uint4*>(l) = *reinterpret_cast<const uint4*>(g);
#endif
}

// ---------------------------------------------------------------------------
// Operand preparation kernels (f32 -> f16, with layout transforms).
// ---------------------------------------------------------------------------
__global__ void cvt_f16_kernel(_Float16* __restrict__ dst, const float* __restrict__ src, size_t n)
{
  size_t i = (size_t)blockIdx.x * 256 + threadIdx.x;
  if (i < n) dst[i] = (_Float16)src[i];
}

// W_in (K=2048 x N=8352) -> f16 transposed [N=8384 (padded)][K=2048]
__global__ void cvt_win_t_kernel(_Float16* __restrict__ dst, const float* __restrict__ src)
{
  size_t i = (size_t)blockIdx.x * 256 + threadIdx.x;
  if (i >= (size_t)DIPP * DMODEL) return;
  int    k = (int)(i % DMODEL);
  size_t n = i / DMODEL;
  dst[i] = (n < DIP) ? (_Float16)src[(size_t)k * DIP + n] : (_Float16)0.f;
}

// W_out (K=4096 x N=2048) -> f16 transposed [N=2048][K=4096]
__global__ void cvt_wout_t_kernel(_Float16* __restrict__ dst, const float* __restrict__ src)
{
  size_t i = (size_t)blockIdx.x * 256 + threadIdx.x;
  if (i >= (size_t)DMODEL * DIN) return;
  int    k = (int)(i % DIN);
  size_t n = i / DIN;
  dst[i] = (_Float16)src[(size_t)k * DMODEL + n];
}

// ---------------------------------------------------------------------------
// GEMM: C[M][N](f32) = A[M][K](f16, row-major) x BT[N][K](f16, K contiguous).
// N % 64 == 0, K % 32 == 0.  Block tile 128(M) x 64(N), 8 waves; wave tile
// 16 x 64 (4 WMMA / k-step).  A (128x32) and BT (64x32) tiles double-buffered
// in LDS via async-to-LDS; all fragment reads are contiguous b128s.
// ---------------------------------------------------------------------------
__global__ __launch_bounds__(256) void gemm_f16_kernel(
    float* __restrict__ C, const _Float16* __restrict__ A, const _Float16* __restrict__ BT,
    int M, int N, int K)
{
  __shared__ __align__(16) _Float16 LA[2][128][40];   // 32 k + 8 pad (80B rows)
  __shared__ __align__(16) _Float16 LBT[2][64][40];   // 32 k + 8 pad (80B rows)
  const int tid   = threadIdx.x;
  const int lane  = tid & 31;
  const int wave  = tid >> 5;
  const int n0    = blockIdx.x * 64;
  const int m0    = blockIdx.y * 128;
  const int r     = lane & 15;
  const int klo   = (lane >> 4) << 3;
  const int kb16  = (lane >> 4) << 4;
  const int ehalf = (lane >> 4) << 3;
  // BT staging map: 256 threads x 16B = 4KB tile (64 rows x 4 chunks)
  const int bnn = tid >> 2;           // 0..63
  const int bk8 = (tid & 3) * 8;      // 0,8,16,24
  // A staging map: 256 threads x 2 chunks x 16B = 8KB tile (128 rows x 4 chunks)
  const int ac0 = tid * 2;

  v8f acc[4];
  #pragma unroll
  for (int q = 0; q < 4; ++q) acc[q] = vzero8();

  auto stage = [&](int buf, int k0){
    copy16_g2l(BT + (size_t)(n0 + bnn) * K + k0 + bk8, &LBT[buf][bnn][bk8]);
    #pragma unroll
    for (int j = 0; j < 2; ++j) {
      int cidx = ac0 + j;
      int row  = cidx >> 2;
      int c8   = (cidx & 3) * 8;
      copy16_g2l(A + (size_t)(m0 + row) * K + k0 + c8, &LA[buf][row][c8]);
    }
  };

  stage(0, 0);
  int buf = 0;
  for (int k0 = 0; k0 < K; k0 += 32) {
    if (k0 + 32 < K) {
      stage(buf ^ 1, k0 + 32);
      asm volatile("s_wait_asynccnt 0x3" ::: "memory");   // current buffer landed
    } else {
      asm volatile("s_wait_asynccnt 0x0" ::: "memory");
    }
    __syncthreads();
    v16h af;
    const _Float16* ar = &LA[buf][wave * 16 + r][0];
    #pragma unroll
    for (int i = 0; i < 8; ++i) { af[i] = ar[kb16 + i]; af[8+i] = ar[kb16 + 8 + i]; }
    #pragma unroll
    for (int q = 0; q < 4; ++q) {
      v16h bf;
      const _Float16* br = &LBT[buf][q * 16 + r][kb16];
      #pragma unroll
      for (int i = 0; i < 16; ++i) bf[i] = br[i];
      acc[q] = wmma16(af, bf, acc[q]);
    }
    __syncthreads();
    buf ^= 1;
  }
  float* cp = C + (size_t)(m0 + wave * 16) * N + n0;
  #pragma unroll
  for (int e = 0; e < 8; ++e) {
    int mm = e + ehalf;
    #pragma unroll
    for (int q = 0; q < 4; ++q) cp[(size_t)mm * N + q * 16 + r] = acc[q][e];
  }
}

// ---------------------------------------------------------------------------
// dt = softplus(zx[..., 8320+h] + dt_bias[h])
// ---------------------------------------------------------------------------
__global__ void dt_kernel(float* __restrict__ dt, const float* __restrict__ zx,
                          const float* __restrict__ dt_bias)
{
  int idx = blockIdx.x * 256 + threadIdx.x;
  if (idx >= BATCHN * SEQ * NH) return;
  int h  = idx & 31;
  int bl = idx >> 5;
  float raw = zx[(size_t)bl * DIPP + (DIN + CONVD) + h] + dt_bias[h];
  dt[idx] = (raw > 20.f) ? raw : log1pf(__expf(raw));
}

// ---------------------------------------------------------------------------
// Causal depthwise conv (K=4) + SiLU over xBC columns of zx.
// ---------------------------------------------------------------------------
__global__ void conv_kernel(float* __restrict__ out, const float* __restrict__ zx,
                            const float* __restrict__ w, const float* __restrict__ bias)
{
  size_t idx = (size_t)blockIdx.x * 256 + threadIdx.x;
  if (idx >= (size_t)BATCHN * SEQ * CONVD) return;
  int    cch  = (int)(idx % CONVD);
  size_t bl   = idx / CONVD;
  int    l    = (int)(bl % SEQ);
  size_t brow = bl - l;   // b*SEQ
  float acc = bias[cch];
  #pragma unroll
  for (int j = 0; j < 4; ++j) {
    int ls = l - 3 + j;
    if (ls >= 0) acc = fmaf(w[cch * 4 + j], zx[(brow + ls) * DIPP + DIN + cch], acc);
  }
  out[idx] = silu_f(acc);
}

// ---------------------------------------------------------------------------
// XsT[b][c][h][p][t] = f16(x * dt)            (t contiguous)
// Bh [b*SEQ + l][n]  = f16(B)                 (n contiguous, for ydiag S-GEMM)
// BhT[b][c][n][t]    = f16(B)                 (t contiguous, for states A-op)
// Ch [b*SEQ + l][n]  = f16(C)
// ---------------------------------------------------------------------------
__global__ __launch_bounds__(256) void prep_kernel(
    _Float16* __restrict__ XsT, _Float16* __restrict__ Bh, _Float16* __restrict__ BhT,
    _Float16* __restrict__ Ch, const float* __restrict__ conv, const float* __restrict__ dt)
{
  const int bl = blockIdx.x, t = threadIdx.x;
  const int l  = bl & (SEQ - 1);
  const int b  = bl >> 12;            // SEQ = 4096
  const int c  = l >> 8;              // CHK = 256
  const int tt = l & 255;
  const float* cp = conv + (size_t)bl * CONVD;
  const float* dp = dt   + (size_t)bl * NH;
  const size_t chbase = (size_t)(b * NC + c) * NH * HD * CHK;
  #pragma unroll
  for (int q = 0; q < 16; ++q) {
    int ch = q * 256 + t;             // ch = h*128 + p
    int h = ch >> 7, pp = ch & 127;
    XsT[chbase + ((size_t)h * HD + pp) * CHK + tt] = (_Float16)(cp[ch] * dp[h]);
  }
  if (t < NSTATE) {
    _Float16 v = (_Float16)cp[DIN + t];
    Bh[(size_t)bl * NSTATE + t] = v;
    BhT[((size_t)(b * NC + c) * NSTATE + t) * CHK + tt] = v;
  } else if (t < 2 * NSTATE) {
    Ch[(size_t)bl * NSTATE + (t - NSTATE)] = (_Float16)cp[DIN + t];
  }
}

// ---------------------------------------------------------------------------
// Acs[b][h][c][t] = inclusive cumsum over t of A[h]*dt  (per chunk).
// ---------------------------------------------------------------------------
__global__ __launch_bounds__(256) void acs_kernel(
    float* __restrict__ Acs, const float* __restrict__ dt, const float* __restrict__ A_log)
{
  __shared__ float s[CHK];
  const int blk = blockIdx.x;           // BATCHN*NH*NC
  const int c  = blk & 15;
  const int bh = blk >> 4;
  const int h  = bh & 31;
  const int b  = bh >> 5;
  const int t  = threadIdx.x;
  const float Aneg = -__expf(A_log[h]);
  size_t l = (size_t)b * SEQ + (size_t)c * CHK + t;
  s[t] = Aneg * dt[l * NH + h];
  __syncthreads();
  for (int off = 1; off < CHK; off <<= 1) {
    float add = (t >= off) ? s[t - off] : 0.f;
    __syncthreads();
    s[t] += add;
    __syncthreads();
  }
  Acs[((size_t)(b * NH + h) * NC + c) * CHK + t] = s[t];
}

// ---------------------------------------------------------------------------
// Y_diag: flash-style.  Per block: one 64-row i-strip of one (b,c,h).
// Phase 1: S = C_i (64x64) @ B_j^T (64x32), mask * exp(Acs_i - Acs_j) -> LDS f16
// Phase 2: Y += S (64x32) @ X_j (32x128)    (X from XsT, K contiguous)
// ---------------------------------------------------------------------------
__global__ __launch_bounds__(256) void ydiag_kernel(
    float* __restrict__ Y, const _Float16* __restrict__ XsT,
    const _Float16* __restrict__ Bh, const _Float16* __restrict__ Ch,
    const float* __restrict__ Acs)
{
  __shared__ _Float16 Sh[64][40];
  const int blk = blockIdx.x;           // BATCHN*NC*NH*4
  const int is = blk & 3;
  const int h  = (blk >> 2) & 31;
  const int c  = (blk >> 7) & 15;
  const int b  = blk >> 11;
  const int lane  = threadIdx.x & 31;
  const int wave  = threadIdx.x >> 5;
  const int r     = lane & 15;
  const int klo   = (lane >> 4) << 3;
  const int kb16  = (lane >> 4) << 4;
  const int ehalf = (lane >> 4) << 3;
  const int i0 = is * 64;
  const size_t lbase = (size_t)b * SEQ + (size_t)c * CHK;
  const float* acsp = Acs + ((size_t)(b * NH + h) * NC + c) * CHK;
  const _Float16* Xb = XsT + ((size_t)(b * NC + c) * NH + h) * HD * CHK;
  const int mt1 = wave >> 1, nt1 = wave & 1;        // phase-1 tile (4x2)
  const int mt2 = wave >> 1;                        // phase-2: 16x64 strip
  const int cb2 = (wave & 1) * 64;
  v8f acc[4];
  #pragma unroll
  for (int q = 0; q < 4; ++q) acc[q] = vzero8();
  const int jbmax = 2 * is + 1;
  for (int jb = 0; jb <= jbmax; ++jb) {
    const int j0 = jb * 32;
    // ---- phase 1 ----
    v8f s = vzero8();
    const _Float16* Cp1 = Ch + (lbase + i0 + mt1 * 16 + r) * NSTATE;
    const _Float16* Bp1 = Bh + (lbase + j0 + nt1 * 16 + r) * NSTATE;
    #pragma unroll
    for (int ks = 0; ks < NSTATE; ks += 32) {
      v16h af, bf;
      #pragma unroll
      for (int i = 0; i < 8; ++i) { af[i] = Cp1[ks + kb16 + i]; af[8+i] = Cp1[ks + kb16 + 8 + i]; }
      #pragma unroll
      for (int i = 0; i < 16; ++i) bf[i] = Bp1[ks + kb16 + i];
      s = wmma16(af, bf, s);
    }
    const float acs_j = acsp[j0 + nt1 * 16 + r];
    __syncthreads();
    #pragma unroll
    for (int e = 0; e < 8; ++e) {
      int mi    = e + ehalf;
      int i_loc = i0 + mt1 * 16 + mi;
      int j_loc = j0 + nt1 * 16 + r;
      float f = (j_loc <= i_loc) ? __expf(acsp[i_loc] - acs_j) : 0.f;
      Sh[mt1 * 16 + mi][nt1 * 16 + r] = (_Float16)(s[e] * f);
    }
    __syncthreads();
    // ---- phase 2 ----
    v16h af2;
    const int srow = mt2 * 16 + r;
    #pragma unroll
    for (int i = 0; i < 8; ++i) { af2[i] = Sh[srow][kb16 + i]; af2[8+i] = Sh[srow][kb16 + 8 + i]; }
    #pragma unroll
    for (int q = 0; q < 4; ++q) {
      v16h bf2;
      const _Float16* xp = Xb + (size_t)(cb2 + q * 16 + r) * CHK + j0 + kb16;
      #pragma unroll
      for (int i = 0; i < 16; ++i) bf2[i] = xp[i];
      acc[q] = wmma16(af2, bf2, acc[q]);
    }
  }
  float* yp = Y + (lbase + i0 + mt2 * 16) * DIN + (size_t)h * HD + cb2;
  #pragma unroll
  for (int e = 0; e < 8; ++e) {
    int mm = e + ehalf;
    #pragma unroll
    for (int q = 0; q < 4; ++q) yp[(size_t)mm * DIN + q * 16 + r] = acc[q][e];
  }
}

// ---------------------------------------------------------------------------
// Chunk states (transposed out):
//   StT[b][c][h][p][n] = sum_t B[t,n] * exp(Acs_last - Acs_t) * X[t,p]
// ---------------------------------------------------------------------------
__global__ __launch_bounds__(256) void states_kernel(
    float* __restrict__ StT, const _Float16* __restrict__ XsT,
    const _Float16* __restrict__ BhT, const float* __restrict__ Acs)
{
  __shared__ float dec[CHK];
  const int blk = blockIdx.x;          // BATCHN*NC*NH
  const int h = blk & 31;
  const int c = (blk >> 5) & 15;
  const int b = blk >> 9;
  const float* acsp = Acs + ((size_t)(b * NH + h) * NC + c) * CHK;
  const float alast = acsp[CHK - 1];
  dec[threadIdx.x] = __expf(alast - acsp[threadIdx.x]);
  __syncthreads();
  const int lane  = threadIdx.x & 31;
  const int wave  = threadIdx.x >> 5;
  const int r     = lane & 15;
  const int klo   = (lane >> 4) << 3;
  const int kb16  = (lane >> 4) << 4;
  const int ehalf = (lane >> 4) << 3;
  const int mt = wave >> 1;
  const int cb = (wave & 1) * 64;
  const int srow = mt * 16 + r;
  const _Float16* Xb = XsT + ((size_t)(b * NC + c) * NH + h) * HD * CHK;
  const _Float16* Bb = BhT + ((size_t)(b * NC + c) * NSTATE + srow) * CHK;
  v8f acc[4];
  #pragma unroll
  for (int q = 0; q < 4; ++q) acc[q] = vzero8();
  for (int t0 = 0; t0 < CHK; t0 += 32) {
    v16h af;
    #pragma unroll
    for (int i = 0; i < 8; ++i) {
      int k1 = t0 + kb16 + i, k2 = t0 + kb16 + 8 + i;
      af[i]   = (_Float16)((float)Bb[k1] * dec[k1]);
      af[8+i] = (_Float16)((float)Bb[k2] * dec[k2]);
    }
    #pragma unroll
    for (int q = 0; q < 4; ++q) {
      v16h bf;
      const _Float16* xp = Xb + (size_t)(cb + q * 16 + r) * CHK + t0 + kb16;
      #pragma unroll
      for (int i = 0; i < 16; ++i) bf[i] = xp[i];
      acc[q] = wmma16(af, bf, acc[q]);
    }
  }
  // element (n = mt*16 + e + ehalf, p = cb + q*16 + r) -> StT[p][n]
  float* sp = StT + (((size_t)(b * NC + c) * NH + h) * HD) * NSTATE + mt * 16;
  #pragma unroll
  for (int q = 0; q < 4; ++q) {
    float* spq = sp + (size_t)(cb + q * 16 + r) * NSTATE + ehalf;
    #pragma unroll
    for (int e = 0; e < 8; ++e) spq[e] = acc[q][e];
  }
}

// ---------------------------------------------------------------------------
// Cross-chunk recurrence, in place (elementwise, layout-agnostic):
//   StT[b][c] <- state entering chunk c.
// ---------------------------------------------------------------------------
__global__ void recur_kernel(float* __restrict__ St, const float* __restrict__ Acs)
{
  const int blk = blockIdx.x;          // BATCHN*NH
  const int h = blk & 31;
  const int b = blk >> 5;
  const int t = threadIdx.x;
  float carry[32];
  #pragma unroll
  for (int k = 0; k < 32; ++k) carry[k] = 0.f;
  for (int c = 0; c < NC; ++c) {
    float a  = Acs[((size_t)(b * NH + h) * NC + c) * CHK + (CHK - 1)];
    float dc = __expf(a);
    float* sp = St + ((size_t)(b * NC + c) * NH + h) * (NSTATE * HD);
    #pragma unroll
    for (int k = 0; k < 32; ++k) {
      size_t e = (size_t)k * 256 + t;
      float s = sp[e];
      sp[e] = carry[k];
      carry[k] = carry[k] * dc + s;
    }
  }
}

// ---------------------------------------------------------------------------
// Y_off: Y += (C (64x64) @ St (64x128)) * exp(Acs_i)   (St from StT[p][n])
// ---------------------------------------------------------------------------
__global__ __launch_bounds__(256) void yoff_kernel(
    float* __restrict__ Y, const _Float16* __restrict__ Ch,
    const float* __restrict__ StT, const float* __restrict__ Acs)
{
  const int blk = blockIdx.x;          // BATCHN*NC*NH*4
  const int is = blk & 3;
  const int h  = (blk >> 2) & 31;
  const int c  = (blk >> 7) & 15;
  const int b  = blk >> 11;
  const int lane  = threadIdx.x & 31;
  const int wave  = threadIdx.x >> 5;
  const int r     = lane & 15;
  const int klo   = (lane >> 4) << 3;
  const int kb16  = (lane >> 4) << 4;
  const int ehalf = (lane >> 4) << 3;
  const int mt = wave >> 1;
  const int cb = (wave & 1) * 64;
  const int i0 = is * 64;
  const float* acsp = Acs + ((size_t)(b * NH + h) * NC + c) * CHK;
  const size_t lbase = (size_t)b * SEQ + (size_t)c * CHK;
  v8f acc[4];
  #pragma unroll
  for (int q = 0; q < 4; ++q) acc[q] = vzero8();
  const _Float16* Cp = Ch + (lbase + i0 + mt * 16 + r) * NSTATE;
  const float* Sp = StT + (((size_t)(b * NC + c) * NH + h) * HD) * NSTATE;
  #pragma unroll
  for (int ks = 0; ks < NSTATE; ks += 32) {
    v16h af;
    #pragma unroll
    for (int i = 0; i < 8; ++i) { af[i] = Cp[ks + kb16 + i]; af[8+i] = Cp[ks + kb16 + 8 + i]; }
    #pragma unroll
    for (int q = 0; q < 4; ++q) {
      v16h bf;
      const float* bp = Sp + (size_t)(cb + q * 16 + r) * NSTATE + ks + kb16;
      #pragma unroll
      for (int i = 0; i < 16; ++i) bf[i] = (_Float16)bp[i];
      acc[q] = wmma16(af, bf, acc[q]);
    }
  }
  float* yp = Y + (lbase + i0 + mt * 16) * DIN + (size_t)h * HD + cb;
  #pragma unroll
  for (int e = 0; e < 8; ++e) {
    int mm = e + ehalf;
    float sc = __expf(acsp[i0 + mt * 16 + mm]);
    #pragma unroll
    for (int q = 0; q < 4; ++q) {
      size_t o = (size_t)mm * DIN + q * 16 + r;
      yp[o] += acc[q][e] * sc;
    }
  }
}

// ---------------------------------------------------------------------------
// Yh = f16( rmsnorm((Y + x*D) * silu(z)) * norm_w )
// ---------------------------------------------------------------------------
__global__ __launch_bounds__(256) void gatenorm_kernel(
    _Float16* __restrict__ Yh, const float* __restrict__ Y,
    const float* __restrict__ conv, const float* __restrict__ zx,
    const float* __restrict__ Dp, const float* __restrict__ nw)
{
  __shared__ float red[256];
  const int bl = blockIdx.x, t = threadIdx.x;
  const float* yp = Y + (size_t)bl * DIN;
  const float* cp = conv + (size_t)bl * CONVD;
  const float* zp = zx + (size_t)bl * DIPP;
  _Float16* op = Yh + (size_t)bl * DIN;
  float yg[16];
  float ss = 0.f;
  #pragma unroll
  for (int q = 0; q < 16; ++q) {
    int ch = q * 256 + t;
    float v = yp[ch] + cp[ch] * Dp[ch >> 7];
    v *= silu_f(zp[ch]);
    yg[q] = v;
    ss += v * v;
  }
  red[t] = ss;
  __syncthreads();
  for (int off = 128; off > 0; off >>= 1) {
    if (t < off) red[t] += red[t + off];
    __syncthreads();
  }
  float scale = rsqrtf(red[0] / (float)DIN + 1e-5f);
  #pragma unroll
  for (int q = 0; q < 16; ++q) {
    int ch = q * 256 + t;
    op[ch] = (_Float16)(yg[q] * scale * nw[ch]);
  }
}

// ---------------------------------------------------------------------------
extern "C" void kernel_launch(void* const* d_in, const int* in_sizes, int n_in,
                              void* d_out, int out_size, void* d_ws, size_t ws_size,
                              hipStream_t stream)
{
  const float* u       = (const float*)d_in[0];
  const float* W_in    = (const float*)d_in[1];
  const float* conv_w  = (const float*)d_in[2];
  const float* conv_b  = (const float*)d_in[3];
  const float* dt_bias = (const float*)d_in[4];
  const float* A_log   = (const float*)d_in[5];
  const float* Dp      = (const float*)d_in[6];
  const float* norm_w  = (const float*)d_in[7];
  const float* W_out   = (const float*)d_in[8];
  float* out = (float*)d_out;

  char* p = (char*)d_ws;
  auto take = [&](size_t bytes) -> char* {
    char* q = p;
    p += (bytes + 255) & ~(size_t)255;
    return q;
  };
  float*    zx    = (float*)   take((size_t)BATCHN * SEQ * DIPP * 4);
  float*    conv  = (float*)   take((size_t)BATCHN * SEQ * CONVD * 4);
  _Float16* XsT   = (_Float16*)take((size_t)BATCHN * SEQ * DIN * 2);
  _Float16* Bhp   = (_Float16*)take((size_t)BATCHN * SEQ * NSTATE * 2);
  _Float16* BhT   = (_Float16*)take((size_t)BATCHN * SEQ * NSTATE * 2);
  _Float16* Chp   = (_Float16*)take((size_t)BATCHN * SEQ * NSTATE * 2);
  float*    dtb   = (float*)   take((size_t)BATCHN * SEQ * NH * 4);
  float*    Acs   = (float*)   take((size_t)BATCHN * NH * NC * CHK * 4);
  float*    StT   = (float*)   take((size_t)BATCHN * NC * NH * NSTATE * HD * 4);
  float*    Y     = (float*)   take((size_t)BATCHN * SEQ * DIN * 4);
  _Float16* uh    = (_Float16*)take((size_t)BATCHN * SEQ * DMODEL * 2);
  _Float16* WinT  = (_Float16*)take((size_t)DIPP * DMODEL * 2);
  _Float16* WoutT = (_Float16*)take((size_t)DMODEL * DIN * 2);
  _Float16* Yh    = (_Float16*)take((size_t)BATCHN * SEQ * DIN * 2);

  const int M = BATCHN * SEQ;   // 8192

  // operand preparation (f32 -> f16, B-operands transposed to K-contiguous)
  {
    size_t nu = (size_t)M * DMODEL;
    cvt_f16_kernel<<<(int)((nu + 255) / 256), 256, 0, stream>>>(uh, u, nu);
    size_t nwi = (size_t)DIPP * DMODEL;
    cvt_win_t_kernel<<<(int)((nwi + 255) / 256), 256, 0, stream>>>(WinT, W_in);
    size_t nwo = (size_t)DMODEL * DIN;
    cvt_wout_t_kernel<<<(int)((nwo + 255) / 256), 256, 0, stream>>>(WoutT, W_out);
  }

  gemm_f16_kernel<<<dim3(DIPP / 64, M / 128), 256, 0, stream>>>(zx, uh, WinT, M, DIPP, DMODEL);
  dt_kernel<<<(M * NH + 255) / 256, 256, 0, stream>>>(dtb, zx, dt_bias);
  conv_kernel<<<(int)(((size_t)M * CONVD + 255) / 256), 256, 0, stream>>>(conv, zx, conv_w, conv_b);
  prep_kernel<<<M, 256, 0, stream>>>(XsT, Bhp, BhT, Chp, conv, dtb);
  acs_kernel<<<BATCHN * NH * NC, 256, 0, stream>>>(Acs, dtb, A_log);
  ydiag_kernel<<<BATCHN * NC * NH * 4, 256, 0, stream>>>(Y, XsT, Bhp, Chp, Acs);
  states_kernel<<<BATCHN * NC * NH, 256, 0, stream>>>(StT, XsT, BhT, Acs);
  recur_kernel<<<BATCHN * NH, 256, 0, stream>>>(StT, Acs);
  yoff_kernel<<<BATCHN * NC * NH * 4, 256, 0, stream>>>(Y, Chp, StT, Acs);
  gatenorm_kernel<<<M, 256, 0, stream>>>(Yh, Y, conv, zx, Dp, norm_w);
  gemm_f16_kernel<<<dim3(DMODEL / 64, M / 128), 256, 0, stream>>>(out, Yh, WoutT, M, DMODEL, DIN);
}
